// DARNN_45749991637072
// MI455X (gfx1250) — matrix operand order repaired and checked by reading, not compiled, not run on hardware
//
#include <hip/hip_runtime.h>
#include <hip/hip_bf16.h>

// ---------------------------------------------------------------------------
// DA-RNN on MI455X (gfx1250): all GEMMs via v_wmma_f32_16x16x32_f16 (f16 in,
// f32 accumulate), state/nonlinear math in f32 VALU, per-step kernels chained
// on one stream.
// ---------------------------------------------------------------------------

typedef _Float16 half8  __attribute__((ext_vector_type(8)));
typedef _Float16 half16 __attribute__((ext_vector_type(16)));
typedef float    v8f    __attribute__((ext_vector_type(8)));

#define Bk  256      // batch
#define TT_ 80       // T+P
#define Nn  96       // drivers
#define Mm  512      // encoder hidden
#define PDd 512      // decoder hidden
#define Td  63       // T-1 decoder steps
#define KE  608      // 96 + 512 fused encoder gate K
#define Pp  16       // horizon

__device__ __forceinline__ float sigf(float x) { return 1.f / (1.f + __expf(-x)); }

__device__ __forceinline__ half16 cmb(half8 lo, half8 hi) {
    half16 r;
#pragma unroll
    for (int i = 0; i < 8; ++i) { r[i] = lo[i]; r[i + 8] = hi[i]; }
    return r;
}

// ---------------------------------------------------------------------------
// Generic WMMA GEMM:  C[rows x N] = A[rows x K] * W[N x K]^T   (f16 x f16 -> f32)
//   block = 128 threads (4 waves); each wave computes a 16 x (16*NTILES) strip.
//   grid.x = rows/64, grid.y = N/(16*NTILES)
//   OUT_MODE 0: f32 raw   1: f32 + bias   2: f16 relu(x + bias)
// ---------------------------------------------------------------------------
template <int NTILES, int OUT_MODE>
__global__ __launch_bounds__(128) void gemm_wmma(
    const _Float16* __restrict__ A, int lda,
    const _Float16* __restrict__ W, int ldw,
    float* __restrict__ Cf, _Float16* __restrict__ Ch, int ldc,
    const float* __restrict__ bias, int K)
{
    const int lane = threadIdx.x & 31;
    const int wave = threadIdx.x >> 5;
    const int g    = lane >> 4;     // lane half-group
    const int mn   = lane & 15;     // A row / W row (C col) within tile
    const long rowTile = (long)blockIdx.x * 4 + wave;
    const int  col0    = blockIdx.y * (NTILES * 16);

    v8f acc[NTILES];
#pragma unroll
    for (int j = 0; j < NTILES; ++j) acc[j] = (v8f){0.f,0.f,0.f,0.f,0.f,0.f,0.f,0.f};

    const _Float16* Arow = A + (rowTile * 16 + mn) * (long)lda;

    for (int kb = 0; kb < K; kb += 32) {
        // A fragment: 16-bit A 16x32 layout (ISA 7.12.2)
        half8 alo = *(const half8*)(Arow + kb + g * 8);
        half8 ahi = *(const half8*)(Arow + kb + 16 + g * 8);
        half16 af = cmb(alo, ahi);
#pragma unroll
        for (int j = 0; j < NTILES; ++j) {
            const _Float16* Wrow = W + (long)(col0 + j * 16 + mn) * ldw + kb + g * 8;
            half8 blo = *(const half8*)(Wrow);
            half8 bhi = *(const half8*)(Wrow + 16);
            half16 bf = cmb(blo, bhi);
            acc[j] = __builtin_amdgcn_wmma_f32_16x16x32_f16(
                false, af, false, bf, (short)0, acc[j], false, false);
        }
    }

#pragma unroll
    for (int j = 0; j < NTILES; ++j) {
        const int col = col0 + j * 16 + mn;
#pragma unroll
        for (int r = 0; r < 8; ++r) {
            const long row = rowTile * 16 + r + 8 * g;
            float x = acc[j][r];
            if (OUT_MODE == 0)      Cf[row * (long)ldc + col] = x;
            else if (OUT_MODE == 1) Cf[row * (long)ldc + col] = x + bias[col];
            else                    Ch[row * (long)ldc + col] =
                                        (_Float16)fmaxf(x + bias[col], 0.f);
        }
    }
}

// ------------------------------ prologue kernels ---------------------------
__global__ void k_f32_to_f16(const float* __restrict__ s, _Float16* __restrict__ d, int n) {
    int i = blockIdx.x * blockDim.x + threadIdx.x;
    if (i < n) d[i] = (_Float16)s[i];
}
__global__ void k_zero_f16(_Float16* __restrict__ d, int n) {
    int i = blockIdx.x * blockDim.x + threadIdx.x;
    if (i < n) d[i] = (_Float16)0.f;
}
__global__ void k_zero_f32(float* __restrict__ d, int n) {
    int i = blockIdx.x * blockDim.x + threadIdx.x;
    if (i < n) d[i] = 0.f;
}
// Wcat[r, 0:96] = W_ih_e[r,:], Wcat[r, 96:608] = W_hh_e[r,:]
__global__ void k_build_wcat(const float* __restrict__ Wih, const float* __restrict__ Whh,
                             _Float16* __restrict__ Wcat) {
    int i = blockIdx.x * blockDim.x + threadIdx.x;
    if (i >= 2048 * KE) return;
    int r = i / KE, k = i % KE;
    float v = (k < Nn) ? Wih[r * Nn + k] : Whh[r * Mm + (k - Nn)];
    Wcat[i] = (_Float16)v;
}
// U_e padded K: Uep[s, 0:80]=U_e[s,:], [80:96]=0
__global__ void k_build_uep(const float* __restrict__ Ue, _Float16* __restrict__ Uep) {
    int i = blockIdx.x * blockDim.x + threadIdx.x;
    if (i >= TT_ * 96) return;
    int s = i / 96, t = i % 96;
    Uep[i] = (_Float16)((t < TT_) ? Ue[s * TT_ + t] : 0.f);
}
// EncT[(b*96+n), t(0:96)] = enc_data[b,t,n] (t<80) else 0
__global__ void k_build_encT(const float* __restrict__ enc, _Float16* __restrict__ AT) {
    int i = blockIdx.x * blockDim.x + threadIdx.x;
    if (i >= Bk * Nn * 96) return;
    int t = i % 96, n = (i / 96) % Nn, b = i / (96 * Nn);
    AT[i] = (_Float16)((t < TT_) ? enc[((long)b * TT_ + t) * Nn + n] : 0.f);
}

// ------------------------------ encoder attention --------------------------
// one block (96 threads) per batch row: e_n = v_e . tanh(q_b + UE_bn),
// softmax over drivers, write xh[b,0:96] = x_t * alpha   (f16)
__global__ void k_enc_attn(const float* __restrict__ qe, const float* __restrict__ UE,
                           const float* __restrict__ v_e, const float* __restrict__ enc,
                           int t, _Float16* __restrict__ xh)
{
    int b = blockIdx.x, n = threadIdx.x;
    const float* q  = qe + b * TT_;
    const float* ue = UE + ((long)b * Nn + n) * TT_;
    float e = 0.f;
#pragma unroll 4
    for (int s = 0; s < TT_; ++s) e += v_e[s] * tanhf(q[s] + ue[s]);
    __shared__ float sm[Nn];
    sm[n] = e; __syncthreads();
    float mx = -1e30f;
    for (int i = 0; i < Nn; ++i) mx = fmaxf(mx, sm[i]);
    float ex = __expf(e - mx);
    __syncthreads(); sm[n] = ex; __syncthreads();
    float sum = 0.f;
    for (int i = 0; i < Nn; ++i) sum += sm[i];
    float alpha = ex / sum;
    float x = enc[((long)b * TT_ + t) * Nn + n];
    xh[b * KE + n] = (_Float16)(x * alpha);
}

// ------------------------------ encoder LSTM pointwise ---------------------
__global__ void k_enc_lstm(const float* __restrict__ gse, const float* __restrict__ bih,
                           const float* __restrict__ bhh, float* __restrict__ ce,
                           _Float16* __restrict__ hc, _Float16* __restrict__ xh,
                           _Float16* __restrict__ Hh, int t)
{
    int idx = blockIdx.x * blockDim.x + threadIdx.x;   // B*M
    int b = idx >> 9, m = idx & 511;
    const float* gt = gse + (long)b * 2048;
    float gi = gt[m]        + bih[m]        + bhh[m];
    float gf = gt[512 + m]  + bih[512 + m]  + bhh[512 + m];
    float gg = gt[1024 + m] + bih[1024 + m] + bhh[1024 + m];
    float go = gt[1536 + m] + bih[1536 + m] + bhh[1536 + m];
    float c = sigf(gf) * ce[idx] + sigf(gi) * tanhf(gg);
    float h = sigf(go) * tanhf(c);
    ce[idx] = c;
    hc[b * 1024 + m]       = (_Float16)h;
    hc[b * 1024 + 512 + m] = (_Float16)c;
    xh[b * KE + Nn + m]    = (_Float16)h;
    Hh[((long)b * TT_ + t) * Mm + m] = (_Float16)h;
}

// ------------------------------ decoder attention --------------------------
// one block (128 threads) per batch row: beta = softmax_t(v_d . tanh(q + UH)),
// ctx = beta . H  -> dc[b,512:1024] (f16); if step>=0 also y_til.
__global__ void k_dec_attn(const float* __restrict__ qd, const float* __restrict__ UH,
                           const float* __restrict__ v_d, const _Float16* __restrict__ Hh,
                           _Float16* __restrict__ dc, const float* __restrict__ wtil,
                           const float* __restrict__ wtil_b, const float* __restrict__ dec,
                           int step, float* __restrict__ ytil)
{
    int b = blockIdx.x, tid = threadIdx.x;
    __shared__ float sm[TT_];
    __shared__ float ctxs[Mm];
    __shared__ float red[128];
    if (tid < TT_) {
        const float* q  = qd + (long)b * Mm;
        const float* uh = UH + ((long)b * TT_ + tid) * Mm;
        float e = 0.f;
#pragma unroll 4
        for (int m = 0; m < Mm; ++m) e += v_d[m] * tanhf(q[m] + uh[m]);
        sm[tid] = e;
    }
    __syncthreads();
    float mx = -1e30f;
    for (int i = 0; i < TT_; ++i) mx = fmaxf(mx, sm[i]);
    float sum = 0.f;
    for (int i = 0; i < TT_; ++i) sum += __expf(sm[i] - mx);
    __syncthreads();
    if (tid < TT_) sm[tid] = __expf(sm[tid] - mx) / sum;
    __syncthreads();
    for (int m = tid; m < Mm; m += 128) {
        float c = 0.f;
        const _Float16* hb = Hh + (long)b * TT_ * Mm + m;
#pragma unroll 4
        for (int t2 = 0; t2 < TT_; ++t2) c += sm[t2] * (float)hb[t2 * Mm];
        ctxs[m] = c;
        dc[b * 1024 + 512 + m] = (_Float16)c;
    }
    __syncthreads();
    if (step >= 0) {
        float p = 0.f;
        for (int m = tid; m < Mm; m += 128) p += wtil[1 + m] * ctxs[m];
        red[tid] = p; __syncthreads();
        if (tid == 0) {
            float s2 = 0.f;
            for (int i = 0; i < 128; ++i) s2 += red[i];
            float y = dec[b * Td + step];
            ytil[b] = wtil[0] * y + s2 + wtil_b[0];
        }
    }
}

// ------------------------------ decoder LSTM pointwise ---------------------
__global__ void k_dec_lstm(const float* __restrict__ gsd, const float* __restrict__ ytil,
                           const float* __restrict__ Wih, const float* __restrict__ bih,
                           const float* __restrict__ bhh, float* __restrict__ sd,
                           _Float16* __restrict__ ds, _Float16* __restrict__ dc)
{
    int idx = blockIdx.x * blockDim.x + threadIdx.x;   // B*PD
    int b = idx >> 9, m = idx & 511;
    const float* gt = gsd + (long)b * 2048;
    float y = ytil[b];
    float gi = gt[m]        + y * Wih[m]        + bih[m]        + bhh[m];
    float gf = gt[512 + m]  + y * Wih[512 + m]  + bih[512 + m]  + bhh[512 + m];
    float gg = gt[1024 + m] + y * Wih[1024 + m] + bih[1024 + m] + bhh[1024 + m];
    float go = gt[1536 + m] + y * Wih[1536 + m] + bih[1536 + m] + bhh[1536 + m];
    float c = sigf(gf) * sd[idx] + sigf(gi) * tanhf(gg);
    float h = sigf(go) * tanhf(c);
    sd[idx] = c;
    ds[b * 1024 + m]       = (_Float16)h;
    ds[b * 1024 + 512 + m] = (_Float16)c;
    dc[b * 1024 + m]       = (_Float16)h;
}

// ---------------------------------------------------------------------------
static inline char* carve(char*& p, size_t bytes) {
    char* r = p;
    p += (bytes + 255) & ~(size_t)255;
    return r;
}

extern "C" void kernel_launch(void* const* d_in, const int* in_sizes, int n_in,
                              void* d_out, int out_size, void* d_ws, size_t ws_size,
                              hipStream_t stream) {
    (void)in_sizes; (void)n_in; (void)out_size; (void)ws_size;
    const float* enc_data = (const float*)d_in[0];
    const float* dec_data = (const float*)d_in[1];
    const float* W_ih_e   = (const float*)d_in[2];
    const float* W_hh_e   = (const float*)d_in[3];
    const float* b_ih_e   = (const float*)d_in[4];
    const float* b_hh_e   = (const float*)d_in[5];
    const float* v_e      = (const float*)d_in[6];
    const float* W_e      = (const float*)d_in[7];
    const float* U_e      = (const float*)d_in[8];
    const float* W_ih_d   = (const float*)d_in[9];
    const float* W_hh_d   = (const float*)d_in[10];
    const float* b_ih_d   = (const float*)d_in[11];
    const float* b_hh_d   = (const float*)d_in[12];
    const float* v_d      = (const float*)d_in[13];
    const float* W_d      = (const float*)d_in[14];
    const float* U_d      = (const float*)d_in[15];
    const float* wtil_w   = (const float*)d_in[16];
    const float* wtil_b   = (const float*)d_in[17];
    const float* lin1_w   = (const float*)d_in[18];
    const float* lin1_b   = (const float*)d_in[19];
    const float* lin2_w   = (const float*)d_in[20];
    const float* lin2_b   = (const float*)d_in[21];
    float* out = (float*)d_out;

    char* p = (char*)d_ws;
    // f16 buffers
    _Float16* Wcat  = (_Float16*)carve(p, (size_t)2048 * KE * 2);
    _Float16* We16  = (_Float16*)carve(p, (size_t)TT_ * 1024 * 2);
    _Float16* Whd16 = (_Float16*)carve(p, (size_t)2048 * PDd * 2);
    _Float16* Wd16  = (_Float16*)carve(p, (size_t)Mm * 1024 * 2);
    _Float16* Ud16  = (_Float16*)carve(p, (size_t)Mm * Mm * 2);
    _Float16* L1w16 = (_Float16*)carve(p, (size_t)PDd * 1024 * 2);
    _Float16* L2w16 = (_Float16*)carve(p, (size_t)Pp * PDd * 2);
    _Float16* Uep   = (_Float16*)carve(p, (size_t)TT_ * 96 * 2);
    _Float16* EncT  = (_Float16*)carve(p, (size_t)Bk * Nn * 96 * 2);
    _Float16* HC    = (_Float16*)carve(p, (size_t)Bk * 1024 * 2);
    _Float16* XH    = (_Float16*)carve(p, (size_t)Bk * KE * 2);
    _Float16* Hh    = (_Float16*)carve(p, (size_t)Bk * TT_ * Mm * 2);
    _Float16* DS    = (_Float16*)carve(p, (size_t)Bk * 1024 * 2);
    _Float16* DC    = (_Float16*)carve(p, (size_t)Bk * 1024 * 2);
    _Float16* O1h   = (_Float16*)carve(p, (size_t)Bk * PDd * 2);
    // f32 buffers
    float* UE  = (float*)carve(p, (size_t)Bk * Nn * TT_ * 4);
    float* QE  = (float*)carve(p, (size_t)Bk * TT_ * 4);
    float* G   = (float*)carve(p, (size_t)Bk * 2048 * 4);
    float* CE  = (float*)carve(p, (size_t)Bk * Mm * 4);
    float* UHf = (float*)carve(p, (size_t)Bk * TT_ * Mm * 4);
    float* QD  = (float*)carve(p, (size_t)Bk * Mm * 4);
    float* SD  = (float*)carve(p, (size_t)Bk * PDd * 4);
    float* YT  = (float*)carve(p, (size_t)Bk * 4);

    auto cvt = [&](const float* s, _Float16* d, int n) {
        k_f32_to_f16<<<(n + 255) / 256, 256, 0, stream>>>(s, d, n);
    };
    // -------- prologue: weight conversion / layout / state init --------
    cvt(W_e,    We16,  TT_ * 1024);
    cvt(W_hh_d, Whd16, 2048 * PDd);
    cvt(W_d,    Wd16,  Mm * 1024);
    cvt(U_d,    Ud16,  Mm * Mm);
    cvt(lin1_w, L1w16, PDd * 1024);
    cvt(lin2_w, L2w16, Pp * PDd);
    k_build_wcat<<<(2048 * KE + 255) / 256, 256, 0, stream>>>(W_ih_e, W_hh_e, Wcat);
    k_build_uep<<<(TT_ * 96 + 255) / 256, 256, 0, stream>>>(U_e, Uep);
    k_build_encT<<<(Bk * Nn * 96 + 255) / 256, 256, 0, stream>>>(enc_data, EncT);
    k_zero_f16<<<(Bk * 1024 + 255) / 256, 256, 0, stream>>>(HC, Bk * 1024);
    k_zero_f16<<<(Bk * KE + 255) / 256, 256, 0, stream>>>(XH, Bk * KE);
    k_zero_f16<<<(Bk * 1024 + 255) / 256, 256, 0, stream>>>(DS, Bk * 1024);
    k_zero_f32<<<(Bk * Mm + 255) / 256, 256, 0, stream>>>(CE, Bk * Mm);
    k_zero_f32<<<(Bk * PDd + 255) / 256, 256, 0, stream>>>(SD, Bk * PDd);

    // -------- UE = (enc^T) @ U_e^T  :  [B*N x 80], K=96 --------
    gemm_wmma<5, 0><<<dim3((Bk * Nn) / 64, 1), 128, 0, stream>>>(
        EncT, 96, Uep, 96, UE, nullptr, TT_, nullptr, 96);

    // -------- encoder scan: 80 steps --------
    for (int t = 0; t < TT_; ++t) {
        // q = [h,c] @ W_e^T : [B x 80], K=1024
        gemm_wmma<5, 0><<<dim3(Bk / 64, 1), 128, 0, stream>>>(
            HC, 1024, We16, 1024, QE, nullptr, TT_, nullptr, 1024);
        k_enc_attn<<<Bk, Nn, 0, stream>>>(QE, UE, v_e, enc_data, t, XH);
        // gates = [x*alpha, h] @ Wcat^T : [B x 2048], K=608
        gemm_wmma<4, 0><<<dim3(Bk / 64, 2048 / 64), 128, 0, stream>>>(
            XH, KE, Wcat, KE, G, nullptr, 2048, nullptr, KE);
        k_enc_lstm<<<(Bk * Mm) / 256, 256, 0, stream>>>(
            G, b_ih_e, b_hh_e, CE, HC, XH, Hh, t);
    }

    // -------- UH = H @ U_d^T : [B*80 x 512], K=512 --------
    gemm_wmma<4, 0><<<dim3((Bk * TT_) / 64, Mm / 64), 128, 0, stream>>>(
        Hh, Mm, Ud16, Mm, UHf, nullptr, Mm, nullptr, Mm);

    // -------- decoder scan: 63 steps --------
    for (int s = 0; s < Td; ++s) {
        // q = [d,s] @ W_d^T : [B x 512], K=1024
        gemm_wmma<4, 0><<<dim3(Bk / 64, Mm / 64), 128, 0, stream>>>(
            DS, 1024, Wd16, 1024, QD, nullptr, Mm, nullptr, 1024);
        k_dec_attn<<<Bk, 128, 0, stream>>>(
            QD, UHf, v_d, Hh, DC, wtil_w, wtil_b, dec_data, s, YT);
        // gates = d @ W_hh_d^T : [B x 2048], K=512 (y_til rank-1 added pointwise)
        gemm_wmma<4, 0><<<dim3(Bk / 64, 2048 / 64), 128, 0, stream>>>(
            DS, 1024, Whd16, PDd, G, nullptr, 2048, nullptr, PDd);
        k_dec_lstm<<<(Bk * PDd) / 256, 256, 0, stream>>>(
            G, YT, W_ih_d, b_ih_d, b_hh_d, SD, DS, DC);
    }

    // -------- final context + MLP head --------
    gemm_wmma<4, 0><<<dim3(Bk / 64, Mm / 64), 128, 0, stream>>>(
        DS, 1024, Wd16, 1024, QD, nullptr, Mm, nullptr, 1024);
    k_dec_attn<<<Bk, 128, 0, stream>>>(
        QD, UHf, v_d, Hh, DC, wtil_w, wtil_b, dec_data, -1, YT);
    // relu([d,ctx] @ lin1^T + b1) -> f16 : [B x 512], K=1024
    gemm_wmma<4, 2><<<dim3(Bk / 64, PDd / 64), 128, 0, stream>>>(
        DC, 1024, L1w16, 1024, nullptr, O1h, PDd, lin1_b, 1024);
    // out = o1 @ lin2^T + b2 : [B x 16], K=512
    gemm_wmma<1, 1><<<dim3(Bk / 64, 1), 128, 0, stream>>>(
        O1h, PDd, L2w16, PDd, out, nullptr, Pp, lin2_b, PDd);
}
